// MemNet_78881369358869
// MI455X (gfx1250) — matrix operand order repaired and checked by reading, not compiled, not run on hardware
//
#include <hip/hip_runtime.h>

#define TT    4096
#define DD    128
#define BB    8192
#define NOUT  1000
#define NOUTP 1008   // padded to a multiple of 16 for WMMA N-tiles
#define BT    16     // batch rows per workgroup
#define NW    8      // waves per block (wave32)
#define THREADS 256

typedef __attribute__((ext_vector_type(16))) __bf16 v16bf;
typedef __attribute__((ext_vector_type(8)))  float  v8f;

#define WMMA_BF16(a, b, c) \
  __builtin_amdgcn_wmma_f32_16x16x32_bf16(false, (a), false, (b), (short)0, (c), false, false)

// ---- fragment loaders (contiguous-k order, same for A and B, wave32) ----------------------

// A-matrix 16x32 bf16: lane m = lane&15; lanes 0-15 hold K 0..15,
// lanes 16-31 hold K 16..31 (pairs packed per VGPR).
__device__ __forceinline__ v16bf load_a_frag(const __bf16* src, int ld, int k0, int lane) {
  int m  = lane & 15;
  int kb = (lane >> 4) << 4;             // 0 or 16
  const __bf16* row = src + m * ld + k0 + kb;
  v16bf a;
#pragma unroll
  for (int i = 0; i < 8; ++i) {
    int kk = 2 * i;
    a[2 * i]     = row[kk];
    a[2 * i + 1] = row[kk + 1];
  }
  return a;
}

// B-matrix 32x16 bf16 where column n is a contiguous row of src (src[n][k]):
// lanes 0-15 hold K=0..15, lanes 16-31 hold K=16..31 -> one 32B vector load.
__device__ __forceinline__ v16bf load_b_frag_rows(const __bf16* src, size_t ld, int nbase,
                                                  int k0, int lane) {
  int n    = nbase + (lane & 15);
  int koff = (lane >> 4) << 4;           // 0 or 16
  return *reinterpret_cast<const v16bf*>(src + (size_t)n * ld + k0 + koff);
}

// ---- pre-pack: history -> bf16 (row-major + transposed), W -> bf16 padded

__global__ void __launch_bounds__(THREADS)
cvt_bf16_kernel(const float* __restrict__ hist, const float* __restrict__ W,
                __bf16* __restrict__ histB, __bf16* __restrict__ histT,
                __bf16* __restrict__ WB) {
  int i = blockIdx.x * blockDim.x + threadIdx.x;
  const int HN = TT * DD;
  const int WN = NOUTP * DD;
  if (i < HN) {
    __bf16 bv = (__bf16)hist[i];
    histB[i] = bv;
    int t = i >> 7, d = i & (DD - 1);
    histT[(size_t)d * TT + t] = bv;      // transposed copy for GEMM2 B-frags
  } else if (i < HN + WN) {
    int j = i - HN;
    int o = j >> 7, d = j & (DD - 1);
    WB[j] = (o < NOUT) ? (__bf16)W[o * DD + d] : (__bf16)0.0f;
  }
}

// ---- fused kernel: one workgroup = 16 batch rows ------------------------

__global__ void __launch_bounds__(THREADS)
memnet_fused_kernel(const float* __restrict__ img, const __bf16* __restrict__ histB,
                    const __bf16* __restrict__ histT, const __bf16* __restrict__ WB,
                    const float* __restrict__ bias, float* __restrict__ out) {
  extern __shared__ char smem[];
  __bf16* memS   = (__bf16*)smem;                          // [BT][TT] bf16 = 128 KB
  float*  scoreS = (float*)smem;                           // reuse: [BT][NOUTP] f32
  __bf16* imgS   = (__bf16*)(smem + BT * TT * 2);          // [BT][DD] bf16 = 4 KB
  __bf16* combS  = (__bf16*)(smem + BT * TT * 2 + 4096);   // [BT][DD] bf16 = 4 KB
  float*  p0     = (float*)(smem + BT * TT * 2 + 8192);    // [16][16] partials
  float*  p1     = p0 + 256;                               // [16][16] partials
  float*  rstat  = p1 + 256;                               // [16][4] row stats

  const int tid  = threadIdx.x;
  const int w    = tid >> 5;    // wave 0..7
  const int lane = tid & 31;
  const int b0   = blockIdx.x * BT;

  // stage img tile (bf16 for WMMA A)
  for (int i = tid; i < BT * DD; i += THREADS)
    imgS[i] = (__bf16)img[(size_t)(b0 + (i >> 7)) * DD + (i & (DD - 1))];
  __builtin_prefetch(histB + (size_t)w * (TT / NW) * DD, 0, 0);  // warm L2
  __syncthreads();

  // ---- Phase 1: mem[16,4096] = img_tile @ hist^T (bf16 WMMA, fp32 acc) --
  {
    v16bf a1[4];
#pragma unroll
    for (int ks = 0; ks < 4; ++ks) a1[ks] = load_a_frag(imgS, DD, ks * 32, lane);
    for (int nt = w; nt < TT / 16; nt += NW) {
      v8f c = {};
#pragma unroll
      for (int ks = 0; ks < 4; ++ks) {
        v16bf b = load_b_frag_rows(histB, DD, nt * 16, ks * 32, lane);
        c = WMMA_BF16(a1[ks], b, c);
      }
      int n = nt * 16 + (lane & 15);
#pragma unroll
      for (int r = 0; r < 8; ++r)
        memS[(r + ((lane >> 4) << 3)) * TT + n] = (__bf16)c[r];
    }
  }
  __syncthreads();

  // ---- Phase 2: per-row L2 norm, softmax stats, weights in-place --------
  // 16 threads per row, each owning a contiguous 256-element chunk so LDS
  // traffic is 32B vector loads (ds_load_b128 pairs), not u16 singles.
  const int rt = tid >> 4;   // row 0..15
  const int ct = tid & 15;   // chunk 0..15
  {
    __bf16* rowp = memS + rt * TT + ct * (TT / 16);
    float ss = 0.0f, mx = -1e30f;
#pragma unroll 4
    for (int j0 = 0; j0 < TT / 16; j0 += 16) {
      v16bf v = *reinterpret_cast<const v16bf*>(rowp + j0);
#pragma unroll
      for (int e = 0; e < 16; ++e) {
        float s = (float)v[e];
        ss += s * s;
        mx = fmaxf(mx, s);
      }
    }
    p0[rt * 16 + ct] = ss;
    p1[rt * 16 + ct] = mx;
    __syncthreads();
    if (tid < 16) {
      float ssr = 0.0f, mr = -1e30f;
      for (int j = 0; j < 16; ++j) {
        ssr += p0[tid * 16 + j];
        mr = fmaxf(mr, p1[tid * 16 + j]);
      }
      rstat[tid * 4 + 0] = 1.0f / sqrtf(ssr);  // 1/||mem_row||
      rstat[tid * 4 + 1] = mr;                 // row max (pre-normalization)
    }
    __syncthreads();
    float invn = rstat[rt * 4 + 0];
    float mr   = rstat[rt * 4 + 1];
    float se = 0.0f;
#pragma unroll 4
    for (int j0 = 0; j0 < TT / 16; j0 += 16) {
      v16bf v = *reinterpret_cast<const v16bf*>(rowp + j0);
#pragma unroll
      for (int e = 0; e < 16; ++e)
        se += __expf(((float)v[e] - mr) * invn);
    }
    p0[rt * 16 + ct] = se;
    __syncthreads();
    if (tid < 16) {
      float s = 0.0f;
      for (int j = 0; j < 16; ++j) s += p0[tid * 16 + j];
      rstat[tid * 4 + 2] = 1.0f / s;
    }
    __syncthreads();
    float invs = rstat[rt * 4 + 2];
#pragma unroll 4
    for (int j0 = 0; j0 < TT / 16; j0 += 16) {
      v16bf v = *reinterpret_cast<const v16bf*>(rowp + j0);
#pragma unroll
      for (int e = 0; e < 16; ++e)
        v[e] = (__bf16)(__expf(((float)v[e] - mr) * invn) * invs);
      *reinterpret_cast<v16bf*>(rowp + j0) = v;
    }
  }
  __syncthreads();

  // ---- Phase 3: hv[16,128] = weights @ hist; combined = hv + img --------
  // B-frags come from the transposed bf16 copy: contiguous 32B loads.
  {
    v8f c = {};
    for (int ks = 0; ks < TT / 32; ++ks) {
      v16bf a = load_a_frag(memS, TT, ks * 32, lane);
      v16bf b = load_b_frag_rows(histT, TT, w * 16, ks * 32, lane);
      c = WMMA_BF16(a, b, c);
    }
    int n = w * 16 + (lane & 15);
#pragma unroll
    for (int r = 0; r < 8; ++r) {
      int m = r + ((lane >> 4) << 3);
      float comb = c[r] + img[(size_t)(b0 + m) * DD + n];
      combS[m * DD + n] = (__bf16)comb;
    }
  }
  __syncthreads();

  // ---- Phase 4: scores[16,1008] = combined @ W^T + bias -----------------
  {
    v16bf a3[4];
#pragma unroll
    for (int ks = 0; ks < 4; ++ks) a3[ks] = load_a_frag(combS, DD, ks * 32, lane);
    for (int nt = w; nt < NOUTP / 16; nt += NW) {
      v8f c = {};
#pragma unroll
      for (int ks = 0; ks < 4; ++ks) {
        v16bf b = load_b_frag_rows(WB, DD, nt * 16, ks * 32, lane);
        c = WMMA_BF16(a3[ks], b, c);
      }
      int n = nt * 16 + (lane & 15);
      float bv = (n < NOUT) ? bias[n] : 0.0f;
#pragma unroll
      for (int r = 0; r < 8; ++r)
        scoreS[(r + ((lane >> 4) << 3)) * NOUTP + n] = c[r] + bv;
    }
  }
  __syncthreads();

  // ---- Phase 5: log_softmax over OUT, write output ----------------------
  {
    float mx = -1e30f;
    for (int j = ct; j < NOUT; j += 16) mx = fmaxf(mx, scoreS[rt * NOUTP + j]);
    p0[rt * 16 + ct] = mx;
    __syncthreads();
    if (tid < 16) {
      float mr = -1e30f;
      for (int j = 0; j < 16; ++j) mr = fmaxf(mr, p0[tid * 16 + j]);
      rstat[tid * 4 + 0] = mr;
    }
    __syncthreads();
    float mr = rstat[rt * 4 + 0];
    float se = 0.0f;
    for (int j = ct; j < NOUT; j += 16) se += __expf(scoreS[rt * NOUTP + j] - mr);
    p1[rt * 16 + ct] = se;
    __syncthreads();
    if (tid < 16) {
      float s = 0.0f;
      for (int j = 0; j < 16; ++j) s += p1[tid * 16 + j];
      rstat[tid * 4 + 1] = mr + __logf(s);   // logsumexp
    }
    __syncthreads();
    float lse = rstat[rt * 4 + 1];
    for (int j = ct; j < NOUT; j += 16)
      out[(size_t)(b0 + rt) * NOUT + j] = scoreS[rt * NOUTP + j] - lse;
  }
}

extern "C" void kernel_launch(void* const* d_in, const int* in_sizes, int n_in,
                              void* d_out, int out_size, void* d_ws, size_t ws_size,
                              hipStream_t stream) {
  const float* hist = (const float*)d_in[0];   // [T, D]
  const float* img  = (const float*)d_in[1];   // [B, D]
  const float* W    = (const float*)d_in[2];   // [OUT, D]
  const float* bias = (const float*)d_in[3];   // [OUT]
  float* out = (float*)d_out;                  // [B, OUT]

  // workspace layout (bf16 packs): histB 1MB | histT 1MB | WB 258KB
  __bf16* histB = (__bf16*)d_ws;
  __bf16* histT = (__bf16*)((char*)d_ws + (size_t)TT * DD * 2);
  __bf16* WB    = (__bf16*)((char*)d_ws + (size_t)2 * TT * DD * 2);

  const int ncvt = TT * DD + NOUTP * DD;
  cvt_bf16_kernel<<<(ncvt + THREADS - 1) / THREADS, THREADS, 0, stream>>>(hist, W, histB,
                                                                          histT, WB);

  // dynamic LDS: 128K score/weight panel + img/comb tiles + reduction scratch
  const size_t smem = (size_t)BT * TT * 2 + 4096 + 4096 + 2 * 256 * 4 + 64 * 4;
  memnet_fused_kernel<<<BB / BT, THREADS, smem, stream>>>(img, histB, histT, WB, bias, out);
}
